// dastar_46162308497519
// MI455X (gfx1250) — matrix operand order repaired and checked by reading, not compiled, not run on hardware
//
#include <hip/hip_runtime.h>

typedef _Float16 v16h __attribute__((ext_vector_type(16)));
typedef float    v8f  __attribute__((ext_vector_type(8)));

#define NB 32
#define HW 1024

// One workgroup = whole problem. 32 wave32s, one wave per batch, lockstep steps
// with a workgroup barrier per step (replicates jnp.all(...) done-coupling).
extern "C" __global__ __launch_bounds__(1024)
void dastar_kernel(const float* __restrict__ cm,
                   const float* __restrict__ sm,
                   const float* __restrict__ gm,
                   float* __restrict__ out,
                   float* __restrict__ ws)
{
  extern __shared__ float smem[];
  float*    g_lds    = smem;                               // NB*HW f32
  float*    open_lds = smem + NB * HW;                     // NB*HW f32
  unsigned* hist     = (unsigned*)(smem + 2 * NB * HW);    // NB*32 bitmask words
  float*    goal_f   = (float*)(hist + NB * 32);           // NB
  int*      goal_i   = (int*)(goal_f + NB);                // NB
  int*      flags    = goal_i + NB;                        // NB

  const int tid  = threadIdx.x;
  const int wave = tid >> 5;
  const int lane = tid & 31;
  float* ws_h   = ws;            // NB*HW heuristic map
  float* ws_par = ws + NB * HW;  // NB*HW parents (float, blended like reference)

  // ---- init g/hist with plain stores ----
  for (int k = 0; k < 32; ++k) g_lds[tid + (k << 10)] = 0.0f;
  if (tid < NB * 32) hist[tid] = 0u;

  // ---- open0 = start_map: async global->LDS DMA (ASYNCcnt path) ----
  // 128KB copy, no VGPR round trip: GLOBAL_LOAD_ASYNC_TO_LDS_B128 x8 per thread.
  {
    unsigned lds_off = (unsigned)(size_t)(void*)open_lds + (unsigned)tid * 128u;
    const float* src = sm + tid * 32;
#pragma unroll
    for (int j = 0; j < 8; ++j) {
      unsigned           lo = lds_off + (unsigned)(j * 16);
      unsigned long long ga = (unsigned long long)(size_t)(src + j * 4);
      asm volatile("global_load_async_to_lds_b128 %0, %1, off"
                   :: "v"(lo), "v"(ga) : "memory");
    }
  }

  // ---- wave 0: goal_loc / goal_idx einsum via V_WMMA_F32_16X16X32_F16 ----
  // D(16x16) = A(16x32) x B(32x16) over 32 K-chunks of the 1024 cells.
  // For K<32: c = 32*ch + K, so A rows have closed form:
  //   row0 = ch, row1 = K, row2 = 32*ch + K  ->  a = w0*ch + w1[slot].
  // All values <= 1023: exact in f16; f32 accumulate is exact.
  if (wave == 0) {
    const int  M  = lane & 15;
    const bool hi = lane >= 16;
    const float w0 = (M == 0) ? 1.0f : ((M == 2) ? 32.0f : 0.0f);
    float w1[16];
    int   Kb[16];
#pragma unroll
    for (int s = 0; s < 16; ++s) {
      // A layout (ISA 7.12.2, 16-bit A 16x32): lanes0-15 K=0..7,16..23;
      // lanes16-31 K=8..15,24..31, packed 2 halves per VGPR.
      int v = s >> 1, p = s & 1;
      int K = (v < 4) ? ((hi ? 8 : 0) + 2 * v + p)
                      : ((hi ? 24 : 16) + 2 * (v - 4) + p);
      w1[s] = (M == 1 || M == 2) ? (float)K : 0.0f;
      // B layout: same K order as A (lanes0-15 K=0..7,16..23; lanes16-31 K=8..15,24..31).
      Kb[s] = (s < 8) ? ((hi ? 8 : 0) + s) : ((hi ? 24 : 16) + (s - 8));
    }
    for (int nt = 0; nt < 2; ++nt) {
      v8f acc = {0.f, 0.f, 0.f, 0.f, 0.f, 0.f, 0.f, 0.f};
      const int brow = (nt << 4) + (lane & 15);   // B column -> batch index
#pragma unroll 1
      for (int ch = 0; ch < 32; ++ch) {
        const float fch = (float)ch;
        v16h A, B;
#pragma unroll
        for (int s = 0; s < 16; ++s) {
          A[s] = (_Float16)(w0 * fch + w1[s]);
          B[s] = (_Float16)gm[brow * HW + (ch << 5) + Kb[s]];
        }
        acc = __builtin_amdgcn_wmma_f32_16x16x32_f16(false, A, false, B,
                                                     (short)0, acc, false, false);
      }
      if (!hi) {  // D VGPR2, lanes 0-15 -> row M=2 = flat goal index
        float gf = acc[2];
        int b = (nt << 4) + M;
        goal_f[b] = gf;
        goal_i[b] = (int)(gf + 0.5f);
      }
    }
  }
  asm volatile("s_wait_asynccnt 0x0" ::: "memory");  // open_lds DMA complete
  __syncthreads();

  // ---- h map (exact reference formula, w=1) and parents0 = goal_idx ----
  for (int k = 0; k < 32; ++k) {
    int cell = tid + (k << 10);
    int b = cell >> 10, c = cell & 1023;
    int gidx = goal_i[b];
    float di = (float)(c >> 5) - (float)(gidx >> 5);
    float dj = (float)(c & 31) - (float)(gidx & 31);
    ws_h[cell]   = sqrtf(di * di + dj * dj) * cm[cell];
    ws_par[cell] = goal_f[b];
  }
  __syncthreads();

  // ---- main lockstep loop: wave b simulates batch b ----
  const int   b    = wave;
  const int   base = b << 10;
  const int   gsel = goal_i[b];
  const float invc = 0.03125f;  // 1/sqrt(1024), exact
  int  tb   = HW - 1;
  bool done = false;
#pragma unroll 1
  for (int t = 0; t < HW; ++t) {
    // argmax of exp(-(g+h)/32)*open, first-index tie-break like jnp.argmax
    float best = -1.0f;
    int   bc   = 0;
#pragma unroll 4
    for (int k = 0; k < 32; ++k) {
      int c = (k << 5) + lane;
      float val = expf(-(g_lds[base + c] + ws_h[base + c]) * invc)
                  * open_lds[base + c];
      if (val > best) { best = val; bc = c; }
    }
    for (int m = 16; m > 0; m >>= 1) {
      float ov = __shfl_xor(best, m, 32);
      int   oc = __shfl_xor(bc,   m, 32);
      if (ov > best || (ov == best && oc < bc)) { best = ov; bc = oc; }
    }
    const int   sel      = bc;                      // all lanes converged
    const int   unsolved = (sel == gsel) ? 0 : 1;   // dist = gm[sel]
    const float g_sel    = g_lds[base + sel];

    if (lane == 0) {
      hist[(b << 5) + (sel >> 5)] |= (1u << (sel & 31));   // hist[sel]=1
      float o = open_lds[base + sel] - (float)unsolved;
      open_lds[base + sel] = fminf(fmaxf(o, 0.0f), 1.0f);
      flags[b] = unsolved;
    }
    // 8 neighbors handled by lanes 0..7 (in-bounds only; conv padding)
    if (lane < 8) {
      int dd = (lane < 4) ? lane : lane + 1;   // skip kernel center
      int di = dd / 3 - 1, dj = dd % 3 - 1;
      int ni = (sel >> 5) + di, nj = (sel & 31) + dj;
      if ((unsigned)ni < 32u && (unsigned)nj < 32u) {
        int c2 = (ni << 5) + nj;
        float kc = (di != 0 && dj != 0) ? 1.4142f : 1.0f;   // SQRT2 constant
        float g2 = g_sel + kc;
        float go = g_lds[base + c2];
        float oo = open_lds[base + c2];
        float hv = (float)((hist[(b << 5) + (c2 >> 5)] >> (c2 & 31)) & 1u);
        float nb = cm[base + c2];                            // neighbor = cm
        float ix = ((1.0f - oo) * (1.0f - hv)
                    + oo * ((go > g2) ? 1.0f : 0.0f)) * nb;  // fractional idx
        g_lds[base + c2]    = g2 * ix + go * (1.0f - ix);
        open_lds[base + c2] = fminf(fmaxf(oo + ix, 0.0f), 1.0f);
        ws_par[base + c2]   = (float)sel * ix + ws_par[base + c2] * (1.0f - ix);
      }
    }
    __syncthreads();                 // flags visible to all batches
    int f = flags[lane];             // lane i reads batch i's unsolved flag
    for (int m = 16; m > 0; m >>= 1) f |= __shfl_xor(f, m, 32);
    __syncthreads();                 // protect flags before next overwrite
    if (f == 0) { done = true; tb = t; break; }   // uniform across WG
  }
  const int current_t = done ? tb : (HW - 1);

  // ---- outputs (per wave, so same-wave store ordering covers WAW) ----
  for (int k = 0; k < 32; ++k) {
    int c = (k << 5) + lane;
    out[base + c] = (float)((hist[(b << 5) + (c >> 5)] >> (c & 31)) & 1u);
    out[NB * HW + base + c] = (c == gsel) ? 1.0f : 0.0f;   // path init = gm
  }
  if (lane == 0) {
    float* po = out + NB * HW + base;
    int loc = (int)ws_par[base + gsel];                    // trunc like astype
    loc = loc < 0 ? 0 : (loc > 1023 ? 1023 : loc);         // JAX clamp gather
#pragma unroll 1
    for (int i = 0; i < current_t; ++i) {
      po[loc] = 1.0f;
      int nl = (int)ws_par[base + loc];
      loc = nl < 0 ? 0 : (nl > 1023 ? 1023 : nl);
    }
  }
}

extern "C" void kernel_launch(void* const* d_in, const int* in_sizes, int n_in,
                              void* d_out, int out_size, void* d_ws, size_t ws_size,
                              hipStream_t stream) {
  const float* cm = (const float*)d_in[0];
  const float* sm = (const float*)d_in[1];
  const float* gm = (const float*)d_in[2];
  // LDS: g + open (256KB) + hist bitmask (4KB) + goal/flags (<1KB) < 320KB WGP
  size_t shmem = (size_t)(2 * NB * HW) * sizeof(float)
               + (size_t)(NB * 32) * sizeof(unsigned)
               + (size_t)NB * (sizeof(float) + 2 * sizeof(int));
  dastar_kernel<<<1, 1024, shmem, stream>>>(cm, sm, gm, (float*)d_out,
                                            (float*)d_ws);
}
